// MultiHeadAttention_163208757853
// MI455X (gfx1250) — compile-verified, matrix operand order checked by reading, not run on hardware
//
#include <hip/hip_runtime.h>
#include <stdint.h>

// ---------- types ----------
typedef _Float16 half_t;
typedef __attribute__((ext_vector_type(16))) _Float16 v16h;
typedef __attribute__((ext_vector_type(8)))  _Float16 v8h;
typedef __attribute__((ext_vector_type(4)))  _Float16 v4h;
typedef __attribute__((ext_vector_type(8)))  float    v8f;

#define DIM      1024
#define SEQ      2048
#define BATCH    4
#define HEADS    16
#define HEAD_DIM 64
#define ROWS     (BATCH * SEQ)       // 8192
#define QKV_LD   (3 * DIM)           // 3072
#define ATTN_SCALE 0.125f            // 1/sqrt(64)

// D = A*B + C, f16 inputs, f32 accumulate
__device__ __forceinline__ v8f wmma16x16x32(v16h a, v16h b, v8f c) {
    return __builtin_amdgcn_wmma_f32_16x16x32_f16(
        /*neg_a=*/false, a, /*neg_b=*/false, b,
        /*c_mod=*/(short)0, c, /*reuse_a=*/false, /*reuse_b=*/false);
}

// 16x32 f16 A/B fragment, per CDNA5 16-bit layout:
// lane l<16: row l, K={0..7,16..23}; lane l>=16: row l-16, K={8..15,24..31}.
__device__ __forceinline__ v16h load_frag(const half_t* __restrict__ base,
                                          int stride, int lane) {
    int r  = lane & 15;
    int k0 = (lane & 16) ? 8 : 0;
    const half_t* p = base + (size_t)r * stride + k0;
    v8h lo = *(const v8h*)(p);
    v8h hi = *(const v8h*)(p + 16);
    return __builtin_shufflevector(lo, hi, 0,1,2,3,4,5,6,7,8,9,10,11,12,13,14,15);
}

// Async global->LDS copy of 16 bytes per lane (tracked by ASYNCcnt).
// lds_off = low 32 bits of the generic LDS pointer (aperture|offset form).
__device__ __forceinline__ void async_copy_b128(unsigned lds_off, const void* gaddr) {
    asm volatile("global_load_async_to_lds_b128 %0, %1, off"
                 :: "v"(lds_off), "v"(gaddr) : "memory");
}
__device__ __forceinline__ void wait_async0() {
    asm volatile("s_wait_asynccnt 0x0" ::: "memory");
}

// ---------- fp32 -> f16 convert ----------
__global__ void __launch_bounds__(256) cvt_f32_to_f16(const float* __restrict__ in,
                                                      half_t* __restrict__ out) {
    size_t i = ((size_t)blockIdx.x * blockDim.x + threadIdx.x) * 4;
    float4 v = *(const float4*)(in + i);
    v4h h = { (half_t)v.x, (half_t)v.y, (half_t)v.z, (half_t)v.w };
    *(v4h*)(out + i) = h;
}

// ---------- WMMA GEMM: C[m,n] = sum_k A[m,k]*B[n,k] (+bias) ----------
// 128x128 tile, 8 waves of 32x64, K-step 32, register-prefetch double buffering.
template<bool OUT_HALF, bool BIAS>
__global__ void __launch_bounds__(256, 1)
gemm_wmma(const half_t* __restrict__ A, const half_t* __restrict__ B,
          const float* __restrict__ bias, void* __restrict__ Cout,
          int M, int N, int K) {
    __shared__ __align__(16) half_t As[128 * 40];   // 80B rows, 16B aligned chunks
    __shared__ __align__(16) half_t Bs[128 * 40];

    const int lane = threadIdx.x & 31;
    const int wave = threadIdx.x >> 5;
    const int bm = blockIdx.x * 128;
    const int bn = blockIdx.y * 128;
    const int wm = (wave & 3) * 32;
    const int wn = (wave >> 2) * 64;

    int rowc[2], kcc[2];
    #pragma unroll
    for (int i = 0; i < 2; ++i) {
        int cc = (int)threadIdx.x * 2 + i;     // 0..511
        rowc[i] = cc >> 2;                      // 0..127
        kcc[i]  = (cc & 3) * 8;                 // 0,8,16,24
    }

    v8f acc[2][4] = {};

    #pragma unroll
    for (int i = 0; i < 2; ++i) {
        *(v8h*)&As[rowc[i] * 40 + kcc[i]] = *(const v8h*)&A[(size_t)(bm + rowc[i]) * K + kcc[i]];
        *(v8h*)&Bs[rowc[i] * 40 + kcc[i]] = *(const v8h*)&B[(size_t)(bn + rowc[i]) * K + kcc[i]];
    }
    __syncthreads();

    for (int k0 = 0; k0 < K; k0 += 32) {
        const bool more = (k0 + 32) < K;
        v8h na[2], nb[2];
        if (more) {                             // overlap next-tile loads with WMMA
            #pragma unroll
            for (int i = 0; i < 2; ++i) {
                na[i] = *(const v8h*)&A[(size_t)(bm + rowc[i]) * K + k0 + 32 + kcc[i]];
                nb[i] = *(const v8h*)&B[(size_t)(bn + rowc[i]) * K + k0 + 32 + kcc[i]];
            }
        }
        v16h af[2], bf[4];
        #pragma unroll
        for (int i = 0; i < 2; ++i) af[i] = load_frag(&As[(wm + i * 16) * 40], 40, lane);
        #pragma unroll
        for (int j = 0; j < 4; ++j) bf[j] = load_frag(&Bs[(wn + j * 16) * 40], 40, lane);
        #pragma unroll
        for (int i = 0; i < 2; ++i)
            #pragma unroll
            for (int j = 0; j < 4; ++j)
                acc[i][j] = wmma16x16x32(af[i], bf[j], acc[i][j]);
        if (more) {
            __syncthreads();
            #pragma unroll
            for (int i = 0; i < 2; ++i) {
                *(v8h*)&As[rowc[i] * 40 + kcc[i]] = na[i];
                *(v8h*)&Bs[rowc[i] * 40 + kcc[i]] = nb[i];
            }
            __syncthreads();
        }
    }

    #pragma unroll
    for (int i = 0; i < 2; ++i) {
        #pragma unroll
        for (int j = 0; j < 4; ++j) {
            int n  = bn + wn + j * 16 + (lane & 15);
            int m0 = bm + wm + i * 16 + ((lane & 16) ? 8 : 0);
            float bv = BIAS ? bias[n] : 0.0f;
            #pragma unroll
            for (int r = 0; r < 8; ++r) {
                float v = acc[i][j][r] + bv;
                if (OUT_HALF)
                    ((half_t*)Cout)[(size_t)(m0 + r) * N + n] = (half_t)v;
                else
                    ((float*)Cout)[(size_t)(m0 + r) * N + n] = v;
            }
        }
    }
}

// ---------- flash attention: one block per (b, h, 128 query rows) ----------
// qkv: [8192,3072] f16 (Q@+0, K@+1024, V@+2048 per row; head offset h*64)
__global__ void __launch_bounds__(256, 1)
attn_wmma(const half_t* __restrict__ qkv, half_t* __restrict__ ctx) {
    const int bh   = blockIdx.y;
    const int b    = bh >> 4;
    const int h    = bh & 15;
    const int q0   = blockIdx.x * 128;
    const int lane = threadIdx.x & 31;
    const int wave = threadIdx.x >> 5;

    __shared__ __align__(16) half_t Ks[128 * 72];      // K tile [key][d]   18.4 KB
    __shared__ __align__(16) half_t Vt[64 * 136];      // V^T   [d][key]    17.4 KB
    __shared__ __align__(16) half_t Ps[8 * 16 * 72];   // per-wave P half   18.4 KB

    const size_t ld = QKV_LD;
    const half_t* Qbase = qkv + ((size_t)(b * SEQ + q0 + wave * 16)) * ld + h * HEAD_DIM;
    v16h qf[2];
    qf[0] = load_frag(Qbase,      (int)ld, lane);
    qf[1] = load_frag(Qbase + 32, (int)ld, lane);

    v8f oacc[4] = {};
    float mrow[8], lrow[8];
    #pragma unroll
    for (int r = 0; r < 8; ++r) { mrow[r] = -3.0e38f; lrow[r] = 0.0f; }

    half_t* Pw = &Ps[wave * 16 * 72];

    for (int kt = 0; kt < SEQ / 128; ++kt) {
        const int kv0 = kt * 128;
        __syncthreads();
        // stage K tile (async DMA to LDS) and transposed V tile (manual)
        #pragma unroll
        for (int i = 0; i < 4; ++i) {
            int cc  = (int)threadIdx.x * 4 + i;   // 0..1023
            int row = cc >> 3;                    // key 0..127
            int kc  = (cc & 7) * 8;               // d chunk
            const half_t* base = qkv + ((size_t)(b * SEQ + kv0 + row)) * ld + h * HEAD_DIM;
            async_copy_b128((unsigned)(uintptr_t)&Ks[row * 72 + kc], base + DIM + kc);
            v8h vv = *(const v8h*)(base + 2 * DIM + kc);
            #pragma unroll
            for (int j = 0; j < 8; ++j) Vt[(kc + j) * 136 + row] = vv[j];
        }
        // prefetch next K/V tile into cache while we compute this one
        if (kt + 1 < SEQ / 128) {
            int prow = (int)threadIdx.x & 127;
            const half_t* nbase = qkv + ((size_t)(b * SEQ + kv0 + 128 + prow)) * ld + h * HEAD_DIM;
            if (threadIdx.x < 128) __builtin_prefetch(nbase + DIM, 0, 1);
            else                   __builtin_prefetch(nbase + 2 * DIM, 0, 1);
        }
        wait_async0();          // K tile landed in LDS
        __syncthreads();

        // S = Q K^T over the whole 128-key tile (16 WMMA / wave)
        v8f sacc[8];
        #pragma unroll
        for (int nt = 0; nt < 8; ++nt) {
            sacc[nt] = (v8f){0,0,0,0,0,0,0,0};
            #pragma unroll
            for (int ks = 0; ks < 2; ++ks)
                sacc[nt] = wmma16x16x32(qf[ks],
                    load_frag(&Ks[(nt * 16) * 72 + ks * 32], 72, lane), sacc[nt]);
        }

        // online softmax (rows 0..7 in lanes 0..15, rows 8..15 in lanes 16..31)
        float mnew[8], corr[8], psum[8];
        #pragma unroll
        for (int r = 0; r < 8; ++r) {
            float rm = sacc[0][r];
            #pragma unroll
            for (int nt = 1; nt < 8; ++nt) rm = fmaxf(rm, sacc[nt][r]);
            #pragma unroll
            for (int mk = 1; mk < 16; mk <<= 1) rm = fmaxf(rm, __shfl_xor(rm, mk, 32));
            rm *= ATTN_SCALE;
            mnew[r] = fmaxf(mrow[r], rm);
            corr[r] = __expf(mrow[r] - mnew[r]);
            mrow[r] = mnew[r];
            psum[r] = 0.0f;
        }
        #pragma unroll
        for (int dt = 0; dt < 4; ++dt)
            #pragma unroll
            for (int r = 0; r < 8; ++r) oacc[dt][r] *= corr[r];

        // P and P*V in two 64-key halves (keeps per-wave P LDS at 16x64)
        #pragma unroll
        for (int hs = 0; hs < 2; ++hs) {
            #pragma unroll
            for (int j = 0; j < 4; ++j) {
                int nt = hs * 4 + j;
                #pragma unroll
                for (int r = 0; r < 8; ++r) {
                    float p = __expf(sacc[nt][r] * ATTN_SCALE - mnew[r]);
                    psum[r] += p;
                    int rr = r + ((lane & 16) ? 8 : 0);
                    Pw[rr * 72 + j * 16 + (lane & 15)] = (half_t)p;
                }
            }
            #pragma unroll
            for (int dt = 0; dt < 4; ++dt) {
                #pragma unroll
                for (int ks = 0; ks < 2; ++ks) {
                    v16h pf = load_frag(&Pw[ks * 32], 72, lane);
                    v16h vf = load_frag(&Vt[(dt * 16) * 136 + hs * 64 + ks * 32], 136, lane);
                    oacc[dt] = wmma16x16x32(pf, vf, oacc[dt]);
                }
            }
        }

        #pragma unroll
        for (int r = 0; r < 8; ++r) {
            float s = psum[r];
            #pragma unroll
            for (int mk = 1; mk < 16; mk <<= 1) s += __shfl_xor(s, mk, 32);
            lrow[r] = lrow[r] * corr[r] + s;
        }
    }

    // normalize + store f16 context for the output projection
    #pragma unroll
    for (int dt = 0; dt < 4; ++dt) {
        #pragma unroll
        for (int r = 0; r < 8; ++r) {
            int m = r + ((lane & 16) ? 8 : 0);
            size_t row = (size_t)(b * SEQ + q0 + wave * 16 + m);
            float o = oacc[dt][r] / lrow[r];
            ctx[row * DIM + h * HEAD_DIM + dt * 16 + (lane & 15)] = (half_t)o;
        }
    }
}

// ---------- host ----------
extern "C" void kernel_launch(void* const* d_in, const int* in_sizes, int n_in,
                              void* d_out, int out_size, void* d_ws, size_t ws_size,
                              hipStream_t stream) {
    (void)in_sizes; (void)n_in; (void)out_size; (void)ws_size;
    const float* x      = (const float*)d_in[0];
    const float* w_qkv  = (const float*)d_in[1];
    const float* w_proj = (const float*)d_in[2];
    const float* b_proj = (const float*)d_in[3];
    float* out = (float*)d_out;

    half_t* xh     = (half_t*)d_ws;                      // 8192*1024
    half_t* wqkvh  = xh     + (size_t)ROWS * DIM;        // 3072*1024
    half_t* wprojh = wqkvh  + (size_t)QKV_LD * DIM;      // 1024*1024
    half_t* qkvh   = wprojh + (size_t)DIM * DIM;         // 8192*3072
    half_t* ctxh   = qkvh   + (size_t)ROWS * QKV_LD;     // 8192*1024

    const int T = 256;
    cvt_f32_to_f16<<<(ROWS * DIM)   / (T * 4), T, 0, stream>>>(x,      xh);
    cvt_f32_to_f16<<<(QKV_LD * DIM) / (T * 4), T, 0, stream>>>(w_qkv,  wqkvh);
    cvt_f32_to_f16<<<(DIM * DIM)    / (T * 4), T, 0, stream>>>(w_proj, wprojh);

    gemm_wmma<true, false><<<dim3(ROWS / 128, QKV_LD / 128), 256, 0, stream>>>(
        xh, wqkvh, nullptr, qkvh, ROWS, QKV_LD, DIM);

    attn_wmma<<<dim3(SEQ / 128, BATCH * HEADS), 256, 0, stream>>>(qkvh, ctxh);

    gemm_wmma<false, true><<<dim3(ROWS / 128, DIM / 128), 256, 0, stream>>>(
        ctxh, wprojh, b_proj, out, ROWS, DIM, DIM);
}
